// BasicBlock_90933047591518
// MI455X (gfx1250) — 8-bit matrix operand order repaired and checked by reading, compiled, not run on hardware
//
#include <hip/hip_runtime.h>
#include <hip/hip_bf16.h>

typedef __attribute__((ext_vector_type(16))) int   v16i;
typedef __attribute__((ext_vector_type(8)))  float v8f;

#define CIN   128
#define COUT  128
#define HHH   56
#define WWW   56
#define HW    3136
#define NB    64
#define NPIX  (NB*HW)
#define BN_EPS 1e-5f

__device__ __forceinline__ unsigned int sign_fp8(float v) {
    return v < 0.0f ? 0xB8u : (v > 0.0f ? 0x38u : 0x00u);
}

union V16 { int4 q[4]; v16i v; };

__global__ __launch_bounds__(128) void prep_channel(
    const float* __restrict__ w, const float* __restrict__ gamma,
    const float* __restrict__ beta, const float* __restrict__ mean,
    const float* __restrict__ var, float* __restrict__ chanA,
    float* __restrict__ chanB, unsigned int* __restrict__ zpage)
{
    int co = threadIdx.x;
    if (co < 32) zpage[co] = 0u;
    if (co >= COUT) return;
    const float* wp = w + (long)co * (CIN * 9);
    float s = 0.0f;
    for (int k = 0; k < CIN * 9; ++k) s += fabsf(wp[k]);
    s *= (1.0f / (float)(CIN * 9));
    float inv = gamma[co] * rsqrtf(var[co] + BN_EPS);
    chanA[co] = s * inv;
    chanB[co] = beta[co] - mean[co] * inv;
}

__global__ __launch_bounds__(256) void prep_wfrag(
    const float* __restrict__ w, unsigned int* __restrict__ wfrag)
{
    int idx = blockIdx.x * 256 + threadIdx.x;
    if (idx >= 9 * 8 * 32 * 16) return;
    int j    = idx & 15;
    int lane = (idx >> 4) & 31;
    int mt   = (idx >> 9) & 7;
    int tap  = idx >> 12;
    int co   = mt * 16 + (lane & 15);
    int koff = (lane >> 4) * 16;
    int ci   = 64 * (j >> 3) + ((j & 2) ? 8 : 0) + ((j & 4) ? 32 : 0)
             + (j & 1) * 4 + koff;
    int kh = tap / 3, kw = tap % 3;
    unsigned int d = 0;
    for (int b = 0; b < 4; ++b) {
        float wv = w[(((long)co * CIN + (ci + b)) * 3 + kh) * 3 + kw];
        d |= sign_fp8(wv) << (8 * b);
    }
    wfrag[idx] = d;
}

__global__ __launch_bounds__(256) void prep_x(
    const float* __restrict__ x, unsigned char* __restrict__ xs8)
{
    __shared__ unsigned char lds[64][32];
    const int t   = threadIdx.x;
    const int hw0 = blockIdx.x * 64;
    const int ci0 = blockIdx.y * 32;
    const int n   = blockIdx.z;

    for (int k = 0; k < 8; ++k) {
        int lin  = t + k * 256;
        int ci_l = lin >> 6;
        int hw_l = lin & 63;
        float v = x[((long)(n * CIN + ci0 + ci_l)) * HW + hw0 + hw_l];
        lds[hw_l][ci_l] = (unsigned char)sign_fp8(v);
    }
    __syncthreads();
    int hw_l = t >> 2;
    int c8   = (t & 3) * 8;
    unsigned int lo = 0, hi = 0;
    for (int b = 0; b < 4; ++b) {
        lo |= ((unsigned int)lds[hw_l][c8 + b])     << (8 * b);
        hi |= ((unsigned int)lds[hw_l][c8 + 4 + b]) << (8 * b);
    }
    uint2 o; o.x = lo; o.y = hi;
    *(uint2*)(xs8 + ((long)(n * HW + hw0 + hw_l)) * CIN + ci0 + c8) = o;
}

__global__ __launch_bounds__(256) void bconv_wmma(
    const unsigned char* __restrict__ xs8,
    const unsigned char* __restrict__ wfrag,
    const float* __restrict__ chanA,
    const float* __restrict__ chanB,
    const unsigned char* __restrict__ zpage,
    const float* __restrict__ xres,
    float* __restrict__ out)
{
    const int lane   = threadIdx.x & 31;
    const int wave   = threadIdx.x >> 5;
    const int wave_n = wave & 3;
    const int wave_m = wave >> 2;
    const int pixbase = blockIdx.x * 128;
    const int lhalf  = lane >> 4;
    const int l15    = lane & 15;

    int pn[2], phw[2];
    bool hv[2][3], wv[2][3];
    const unsigned char* bctr[2];
#pragma unroll
    for (int jn = 0; jn < 2; ++jn) {
        int pix = pixbase + wave_n * 32 + jn * 16 + l15;
        int n   = pix / HW;
        int hw  = pix - n * HW;
        int h   = hw / WWW;
        int w   = hw - h * WWW;
        pn[jn]  = n;
        phw[jn] = hw;
        hv[jn][0] = (h > 0);        hv[jn][1] = true; hv[jn][2] = (h < HHH - 1);
        wv[jn][0] = (w > 0);        wv[jn][1] = true; wv[jn][2] = (w < WWW - 1);
        bctr[jn] = xs8 + ((long)n * HW + hw) * CIN + lhalf * 16;
    }

    const unsigned char* abase = wfrag + (long)wave_m * 8192 + lane * 64;

    v8f acc[4][2];
#pragma unroll
    for (int i = 0; i < 4; ++i)
#pragma unroll
        for (int jn = 0; jn < 2; ++jn)
            acc[i][jn] = (v8f)0.0f;

#pragma unroll
    for (int tap = 0; tap < 9; ++tap) {
        const int dh = tap / 3;
        const int dw = tap % 3;
        const int disp = ((dh - 1) * WWW + (dw - 1)) * CIN;

        V16 Bf[2];
#pragma unroll
        for (int jn = 0; jn < 2; ++jn) {
            bool valid = hv[jn][dh] && wv[jn][dw];
            const unsigned char* src = valid ? (bctr[jn] + disp) : zpage;
#pragma unroll
            for (int g = 0; g < 4; ++g)
                Bf[jn].q[g] = *(const int4*)(src + 32 * g);
        }

        V16 Af[4];
#pragma unroll
        for (int i = 0; i < 4; ++i) {
#pragma unroll
            for (int g = 0; g < 4; ++g)
                Af[i].q[g] = *(const int4*)(abase + tap * 16384 + i * 2048 + g * 16);
        }

#pragma unroll
        for (int i = 0; i < 4; ++i)
#pragma unroll
            for (int jn = 0; jn < 2; ++jn)
                acc[i][jn] = __builtin_amdgcn_wmma_f32_16x16x128_fp8_fp8(
                    Af[i].v, Bf[jn].v, (short)0, acc[i][jn], false, false);
    }

    const int co0 = wave_m * 64 + 8 * lhalf;
    const float* aptr = chanA + co0;
    const float* bptr = chanB + co0;
    const float* rp[2];
    float* op[2];
#pragma unroll
    for (int jn = 0; jn < 2; ++jn) {
        long base = ((long)pn[jn] * COUT + co0) * HW + phw[jn];
        rp[jn] = xres + base;
        op[jn] = out + base;
    }
#pragma unroll
    for (int i = 0; i < 4; ++i) {
#pragma unroll
        for (int r = 0; r < 8; ++r) {
            const int c = i * 16 + r;
            const float a = aptr[c];
            const float b = bptr[c];
#pragma unroll
            for (int jn = 0; jn < 2; ++jn) {
                float res = __builtin_nontemporal_load(rp[jn] + (long)c * HW);
                __builtin_nontemporal_store(acc[i][jn][r] * a + b + res,
                                            op[jn] + (long)c * HW);
            }
        }
    }
}

extern "C" void kernel_launch(void* const* d_in, const int* in_sizes, int n_in,
                              void* d_out, int out_size, void* d_ws, size_t ws_size,
                              hipStream_t stream) {
    const float* x      = (const float*)d_in[0];
    const float* weight = (const float*)d_in[1];
    const float* gamma  = (const float*)d_in[2];
    const float* beta   = (const float*)d_in[3];
    const float* bmean  = (const float*)d_in[4];
    const float* bvar   = (const float*)d_in[5];
    float* out = (float*)d_out;

    unsigned char* xs8   = (unsigned char*)d_ws;
    unsigned char* wfrag = xs8 + (size_t)NPIX * CIN;
    float* chanA         = (float*)(wfrag + 9 * 8 * 32 * 64);
    float* chanB         = chanA + COUT;
    unsigned char* zpage = (unsigned char*)(chanB + COUT);

    prep_channel<<<1, 128, 0, stream>>>(weight, gamma, beta, bmean, bvar,
                                        chanA, chanB, (unsigned int*)zpage);
    prep_wfrag<<<144, 256, 0, stream>>>(weight, (unsigned int*)wfrag);
    prep_x<<<dim3(49, 4, 64), 256, 0, stream>>>(x, xs8);
    bconv_wmma<<<NPIX / 128, 256, 0, stream>>>(xs8, wfrag, chanA, chanB, zpage,
                                               x, out);
}
